// PlainRNN_48533130444803
// MI455X (gfx1250) — matrix operand order repaired and checked by reading, not compiled, not run on hardware
//
#include <hip/hip_runtime.h>
#include <hip/hip_bf16.h>
#include <math.h>

// Problem constants (match reference).
constexpr int kB  = 32;
constexpr int kT  = 2048;
constexpr int kNI = 128;
constexpr int kNH = 512;
constexpr int kNO = 128;

// LDS row pitches (elements), padded to spread banks, rows stay 16B aligned.
constexpr int kXP = kNH + 8;   // 520 -> 1040B rows
constexpr int kUP = kNI + 8;   // 136 ->  272B rows

// Dynamic LDS layout (bytes).
constexpr int kXbufBytes  = 2 * 16 * kXP * 2;            // 33280
constexpr int kUbufBytes  = 2 * 16 * kUP * 2;            // 8704
constexpr int kUbufOff    = kXbufBytes;                  // 33280
constexpr int kWrOddOff   = kXbufBytes + kUbufBytes;     // 41984 (16B aligned)
constexpr int kWrChunk    = 32 * 32;                     // 1KB per (wave,kb): 32 lanes x 32B
constexpr int kWrOddBytes = 16 * 16 * kWrChunk;          // 262144
constexpr int kWrE15Off   = kWrOddOff + kWrOddBytes;     // 304128
constexpr int kWrE15Bytes = 16 * kWrChunk;               // 16384 (1 chunk per wave)
constexpr int kSmemBytes  = kWrE15Off + kWrE15Bytes;     // 320512 <= 327680

typedef __bf16 bf16_t;
typedef bf16_t          v16bf __attribute__((ext_vector_type(16)));
typedef float           v8f   __attribute__((ext_vector_type(8)));
typedef float           v4f   __attribute__((ext_vector_type(4)));
typedef unsigned short  v16us __attribute__((ext_vector_type(16)));

union BF16x16 {
    v16us us;
    v16bf bf;
    uint4 q[2];
};

__device__ __forceinline__ unsigned short f32_to_bf16(float f) {
    unsigned int u = __float_as_uint(f);
    unsigned int r = u + 0x7FFFu + ((u >> 16) & 1u);   // round-to-nearest-even
    return (unsigned short)(r >> 16);
}

__device__ __forceinline__ float fast_tanh(float x) {
#if __has_builtin(__builtin_amdgcn_tanhf)
    return __builtin_amdgcn_tanhf(x);        // v_tanh_f32 (gfx1250 trans op)
#elif __has_builtin(__builtin_amdgcn_tanh_f32)
    return __builtin_amdgcn_tanh_f32(x);
#else
    return tanhf(x);
#endif
}

// A operand (16x32 bf16) from row-major bf16 LDS tile.
// Contiguous k order used here: lane<16 -> row=lane,    K = 0..15
//             lane>=16 -> row=lane-16, K = 16..31
__device__ __forceinline__ v16bf load_a_tile(const unsigned short* row0,
                                             int lane, int pitch) {
    const unsigned short* p = row0 + (lane & 15) * pitch + ((lane >> 4) << 4);
    BF16x16 t;
    t.q[0] = *(const uint4*)(p);
    t.q[1] = *(const uint4*)(p + 8);
    return t.bf;
}

// B operand (32x16 bf16) from f32 weight rows: B[k][n] = W[row0+n][k0+k].
// Contiguous k order used here: lanes 0-15 -> col n, K = 0..15; lanes 16-31 -> col n-16, K = 16..31.
__device__ __forceinline__ BF16x16 make_b_from_f32(const float* __restrict__ W,
                                                   int ld, int row0, int k0,
                                                   int lane) {
    const float* p = W + (size_t)(row0 + (lane & 15)) * ld + k0 + ((lane >> 4) << 4);
    BF16x16 t;
#pragma unroll
    for (int i = 0; i < 16; ++i) t.us[i] = f32_to_bf16(p[i]);
    return t;
}

// B operand reload from LDS (stored in exact operand layout, 32B per lane).
__device__ __forceinline__ v16bf load_b_lds(const char* base, int lane) {
    const char* p = base + lane * 32;
    BF16x16 t;
    t.q[0] = *(const uint4*)(p);
    t.q[1] = *(const uint4*)(p + 16);
    return t.bf;
}

__device__ __forceinline__ v8f wmma_bf16(v16bf a, v16bf b, v8f c) {
    return __builtin_amdgcn_wmma_f32_16x16x32_bf16(
        /*neg_a=*/false, a, /*neg_b=*/false, b,
        /*c_mod=*/(short)0, c, /*reuse_a=*/false, /*reuse_b=*/false);
}

// Stage u[b0..b0+15, t, 0..127] (f32) -> bf16 LDS strip. 512 threads:
// tid>>5 = row (16 rows), (tid&31)*4 = column quad -> NT float4 load, 8B LDS store.
__device__ __forceinline__ void stage_u(const float* __restrict__ u,
                                        unsigned short* dst, int b0, int t,
                                        int tid) {
    const int r = tid >> 5;
    const int c = (tid & 31) << 2;
    const v4f v = __builtin_nontemporal_load(
        (const v4f*)(u + ((size_t)(b0 + r) * kT + t) * kNI + c));
    uint2 pk;
    pk.x = (unsigned int)f32_to_bf16(v.x) | ((unsigned int)f32_to_bf16(v.y) << 16);
    pk.y = (unsigned int)f32_to_bf16(v.z) | ((unsigned int)f32_to_bf16(v.w) << 16);
    *(uint2*)(dst + r * kUP + c) = pk;
}

__global__ __launch_bounds__(512, 1)
void PlainRNN_scan_kernel(const float* __restrict__ u,
                          const float* __restrict__ Wr,
                          const float* __restrict__ Wi,
                          const float* __restrict__ Wo,
                          float* __restrict__ z,
                          float* __restrict__ X,
                          float* __restrict__ As,
                          float* __restrict__ Phis) {
    extern __shared__ char smem[];
    unsigned short* xbuf  = (unsigned short*)(smem);              // [2][16*kXP]
    unsigned short* ubuf  = (unsigned short*)(smem + kUbufOff);   // [2][16*kUP]
    char*           wrOdd = smem + kWrOddOff;                     // [16 waves][16 kb][1KB]
    char*           wrE15 = smem + kWrE15Off;                     // [16 waves][1KB]

    const int tid  = threadIdx.x;
    const int lane = tid & 31;
    const int wv   = tid >> 5;          // wave id 0..15
    const int b0   = blockIdx.x * 16;   // batch tile base (2 blocks)
    const int n0e  = (2 * wv) * 16;     // even N-tile column base
    const int n0o  = n0e + 16;          // odd  N-tile column base
    const char* wrOddBase = wrOdd + wv * 16 * kWrChunk;
    const char* wrE15Base = wrE15 + wv * kWrChunk;

    // ---- Even tile: pin Wr k-blocks 0..14 + Wi in VGPRs; kb=15 goes to LDS ----
    v16bf wrE[15];
#pragma unroll
    for (int kb = 0; kb < 15; ++kb)
        wrE[kb] = make_b_from_f32(Wr, kNH, n0e, kb * 32, lane).bf;
    {
        BF16x16 t = make_b_from_f32(Wr, kNH, n0e, 15 * 32, lane);
        char* dst = wrE15 + wv * kWrChunk + lane * 32;
        *(uint4*)(dst)      = t.q[0];
        *(uint4*)(dst + 16) = t.q[1];
    }
    v16bf wiE[4];
#pragma unroll
    for (int kb = 0; kb < 4; ++kb)
        wiE[kb] = make_b_from_f32(Wi, kNI, n0e, kb * 32, lane).bf;

    // ---- Odd tile: Wi pinned; Wr staged once into per-wave LDS (B layout) ----
    v16bf wiO[4];
#pragma unroll
    for (int kb = 0; kb < 4; ++kb)
        wiO[kb] = make_b_from_f32(Wi, kNI, n0o, kb * 32, lane).bf;
#pragma unroll
    for (int kb = 0; kb < 16; ++kb) {
        BF16x16 t = make_b_from_f32(Wr, kNH, n0o, kb * 32, lane);
        char* dst = wrOdd + (wv * 16 + kb) * kWrChunk + lane * 32;
        *(uint4*)(dst)      = t.q[0];
        *(uint4*)(dst + 16) = t.q[1];
    }

    // ---- Init: x0 = 0; stage u[t=0] ----
    for (int i = tid; i < 16 * kXP; i += 512) {
        xbuf[i]            = 0;   // buffer 0
        xbuf[16 * kXP + i] = 0;   // buffer 1
    }
    stage_u(u, ubuf, b0, 0, tid);
    __syncthreads();

    const int cl = lane & 15;            // C/D column within tile
    const int mh = (lane >> 4) << 3;     // C/D row-half offset (0 or 8)

    // ---- Sequential scan: one barrier per step, 40 bf16 WMMAs per wave ----
    for (int t = 0; t < kT; ++t) {
        const int par = t & 1;
        const int nxt = par ^ 1;
        const unsigned short* xb = xbuf + par * 16 * kXP;
        const unsigned short* ub = ubuf + par * 16 * kUP;
        unsigned short*       xn = xbuf + nxt * 16 * kXP;

        // Prefetch next timestep's input while we compute this one.
        if (t + 1 < kT) stage_u(u, ubuf + nxt * 16 * kUP, b0, t + 1, tid);

        // ---------- even N-tile (software-pipelined A loads) ----------
        v16bf a0 = load_a_tile(ub, lane, kUP);
        {
            v8f acc = {0.f, 0.f, 0.f, 0.f, 0.f, 0.f, 0.f, 0.f};
#pragma unroll
            for (int kb = 0; kb < 4; ++kb) {
                v16bf a1 = (kb < 3) ? load_a_tile(ub + (kb + 1) * 32, lane, kUP)
                                    : load_a_tile(xb, lane, kXP);
                acc = wmma_bf16(a0, wiE[kb], acc);
                a0 = a1;
            }
            v16bf bE15;
#pragma unroll
            for (int kb = 0; kb < 15; ++kb) {
                if (kb == 13) bE15 = load_b_lds(wrE15Base, lane);  // depth-2 slack
                v16bf a1 = load_a_tile(xb + (kb + 1) * 32, lane, kXP);
                acc = wmma_bf16(a0, wrE[kb], acc);
                a0 = a1;
            }
            {   // kb = 15 with LDS-resident B; prefetch odd pass start
                v16bf a1 = load_a_tile(ub, lane, kUP);
                acc = wmma_bf16(a0, bE15, acc);
                a0 = a1;
            }
            // Epilogue: publish x_t to LDS first (critical path), stream globals NT.
#pragma unroll
            for (int r = 0; r < 8; ++r) {
                const int m = mh + r;
                const size_t o = ((size_t)(b0 + m) * kT + t) * kNH + n0e + cl;
                const float av = acc[r];
                const float ph = fast_tanh(av);
                xn[m * kXP + n0e + cl] = f32_to_bf16(ph);
                __builtin_nontemporal_store(av, As + o);
                __builtin_nontemporal_store(ph, X + o);
                __builtin_nontemporal_store(ph, Phis + o);
            }
        }
        // ---------- odd N-tile (A and LDS-B both software-pipelined) ----------
        {
            v8f acc = {0.f, 0.f, 0.f, 0.f, 0.f, 0.f, 0.f, 0.f};
            v16bf bcur = load_b_lds(wrOddBase, lane);   // x-part chunk 0, in flight early
#pragma unroll
            for (int kb = 0; kb < 4; ++kb) {
                v16bf a1 = (kb < 3) ? load_a_tile(ub + (kb + 1) * 32, lane, kUP)
                                    : load_a_tile(xb, lane, kXP);
                acc = wmma_bf16(a0, wiO[kb], acc);
                a0 = a1;
            }
#pragma unroll
            for (int kb = 0; kb < 16; ++kb) {
                v16bf b1 = bcur;
                if (kb < 15) b1 = load_b_lds(wrOddBase + (kb + 1) * kWrChunk, lane);
                v16bf a1 = a0;
                if (kb < 15) a1 = load_a_tile(xb + (kb + 1) * 32, lane, kXP);
                acc = wmma_bf16(a0, bcur, acc);
                a0 = a1;
                bcur = b1;
            }
#pragma unroll
            for (int r = 0; r < 8; ++r) {
                const int m = mh + r;
                const size_t o = ((size_t)(b0 + m) * kT + t) * kNH + n0o + cl;
                const float av = acc[r];
                const float ph = fast_tanh(av);
                xn[m * kXP + n0o + cl] = f32_to_bf16(ph);
                __builtin_nontemporal_store(av, As + o);
                __builtin_nontemporal_store(ph, X + o);
                __builtin_nontemporal_store(ph, Phis + o);
            }
        }
        __syncthreads();
    }

    // ---- z = x_T @ Wo^T (x_T sits in buffer 0; waves 0..7, one N-tile each) ----
    if (wv < kNO / 16) {   // wave-uniform branch: EXEC stays all-ones for WMMA
        v8f acc = {0.f, 0.f, 0.f, 0.f, 0.f, 0.f, 0.f, 0.f};
#pragma unroll
        for (int kb = 0; kb < 16; ++kb) {
            v16bf a  = load_a_tile(xbuf + kb * 32, lane, kXP);
            v16bf bm = make_b_from_f32(Wo, kNH, wv * 16, kb * 32, lane).bf;
            acc = wmma_bf16(a, bm, acc);
        }
#pragma unroll
        for (int r = 0; r < 8; ++r)
            z[(size_t)(b0 + mh + r) * kNO + wv * 16 + cl] = acc[r];
    }
}

extern "C" void kernel_launch(void* const* d_in, const int* in_sizes, int n_in,
                              void* d_out, int out_size, void* d_ws, size_t ws_size,
                              hipStream_t stream) {
    (void)in_sizes; (void)n_in; (void)out_size; (void)d_ws; (void)ws_size;

    const float* u  = (const float*)d_in[0];   // [B, T, NI]
    const float* Wr = (const float*)d_in[1];   // [NH, NH]
    const float* Wi = (const float*)d_in[2];   // [NH, NI]
    const float* Wo = (const float*)d_in[3];   // [NO, NH]

    float* out  = (float*)d_out;
    float* z    = out;                                   // [B, NO]
    float* X    = z  + (size_t)kB * kNO;                 // [B, T, NH]
    float* As   = X  + (size_t)kB * kT * kNH;            // [B, T, NH]
    float* Phis = As + (size_t)kB * kT * kNH;            // [B, T, NH] (== X)

    PlainRNN_scan_kernel<<<dim3(2), dim3(512), kSmemBytes, stream>>>(
        u, Wr, Wi, Wo, z, X, As, Phis);
}
